// LSTMEncoder_8890582302973
// MI455X (gfx1250) — matrix operand order repaired and checked by reading, not compiled, not run on hardware
//
#include <hip/hip_runtime.h>
#include <hip/hip_bf16.h>
#include <stdint.h>

// ---------------------------------------------------------------------------
// LSTM encoder for MI455X (gfx1250).
//   B=64, T=512, F=128, H=256, L=128, gate order i,f,g,o.
// Strategy:
//   * time-major bf16 activations; dropout folded into producer stores
//   * projection GEMMs: v_wmma_f32_16x16x32_bf16, A/B as 2x b128 loads,
//     output written in WMMA C-fragment order for the recurrence
//   * recurrence: persistent kernel, cell state in VGPRs for all 512 steps;
//     recurrent weights parked in LDS once via TDM; h broadcast each step via
//     TDM tensor_load_to_lds (double-buffered); grid barrier per step
// ---------------------------------------------------------------------------

typedef __attribute__((ext_vector_type(16))) __bf16 v16bf;
typedef __attribute__((ext_vector_type(8)))  float  v8f;
typedef unsigned __attribute__((ext_vector_type(4))) v4u;
typedef int __attribute__((ext_vector_type(8)))      v8i;
typedef int __attribute__((ext_vector_type(4)))      v4i;

union Frag16 {            // one WMMA 16-bit operand: two contiguous 16B chunks
    v16bf v;
    uint4 q[2];
};

__device__ __forceinline__ float sigmoidf_(float x) {
    return 1.0f / (1.0f + __expf(-x));
}

__device__ __forceinline__ v8f wmma_bf16(v16bf a, v16bf b, v8f c) {
    return __builtin_amdgcn_wmma_f32_16x16x32_bf16(
        false, a, false, b, (short)0, c, false, false);
}

// Grid-wide barrier (all workgroups of the persistent kernel co-resident).
__device__ __forceinline__ void gridbar(unsigned* bar, unsigned target) {
    __syncthreads();
    if (threadIdx.x == 0) {
        __threadfence();
        atomicAdd(bar, 1u);
        while (__hip_atomic_load(bar, __ATOMIC_RELAXED, __HIP_MEMORY_SCOPE_AGENT) < target) {
            __builtin_amdgcn_s_sleep(1);
        }
        __threadfence();
    }
    __syncthreads();
}

// Build a 2-D TDM descriptor: tile = dim0 cols x rows rows of bf16,
// row stride == dim0 (densely packed slice starting at gaddr).
__device__ __forceinline__ void tdm_desc(v4u& g0, v8i& g1, unsigned lds_off,
                                         const void* gaddr, int dim0, int rows) {
    unsigned long long ga = (unsigned long long)(uintptr_t)gaddr;
    g0[0] = 1u;                                              // count=1 (valid)
    g0[1] = lds_off;                                         // lds_addr
    g0[2] = (unsigned)(ga & 0xFFFFFFFFu);                    // global_addr[31:0]
    g0[3] = (unsigned)((ga >> 32) & 0x01FFFFFFu) | (2u << 30);   // [56:32] | type=2
    g1[0] = 1 << 16;                                         // data_size = 2 bytes
    g1[1] = (int)((dim0 & 0xFFFF) << 16);                    // tensor_dim0 lo16
    g1[2] = (int)(((unsigned)dim0 >> 16) & 0xFFFFu) | (int)((rows & 0xFFFF) << 16);
    g1[3] = (int)(((unsigned)rows >> 16) & 0xFFFFu) | (int)((dim0 & 0xFFFF) << 16); // tile_dim0
    g1[4] = rows & 0xFFFF;                                   // tile_dim1 | tile_dim2=0
    g1[5] = dim0;                                            // tensor_dim0_stride lo32
    g1[6] = 0;
    g1[7] = 0;
}

// Tensor Data Mover: issue one 2-D tile load described by D# groups 0/1.
__device__ __forceinline__ void tdm_load(v4u g0, v8i g1) {
    v4i z4 = {0, 0, 0, 0};
#if defined(__clang_major__) && (__clang_major__ >= 23)
    v8i z8 = {0, 0, 0, 0, 0, 0, 0, 0};
    __builtin_amdgcn_tensor_load_to_lds(g0, g1, z4, z4, z8, 0);
#else
    __builtin_amdgcn_tensor_load_to_lds(g0, g1, z4, z4, 0);
#endif
}

// ---------------------------------------------------------------------------
// Wt[n*K + k] = (bf16) W[k*N + n]        (input-projection weights)
// ---------------------------------------------------------------------------
__global__ void k_transpose_bf16(const float* __restrict__ W, __bf16* __restrict__ Wt,
                                 int K, int N) {
    int i = blockIdx.x * blockDim.x + threadIdx.x;
    if (i >= K * N) return;
    int n = i / K, k = i % K;
    Wt[i] = (__bf16)W[(size_t)k * N + n];
}

// ---------------------------------------------------------------------------
// Recurrent weights: U[HN, 4*HN] -> Ut2 transposed bf16, rows reordered as
// (jt, gate, col) so each workgroup's slice (2 jt values) is one contiguous
// 128-row x HN tile for the TDM.
//   Ut2[((jt*4 + g)*16 + c)*HN + k] = U[k*4HN + g*HN + jt*16 + c]
// ---------------------------------------------------------------------------
__global__ void k_transpose_ut(const float* __restrict__ U, __bf16* __restrict__ Ut2,
                               int HN) {
    int i = blockIdx.x * blockDim.x + threadIdx.x;
    if (i >= HN * 4 * HN) return;
    int k  = i % HN;
    int np = i / HN;               // reordered row
    int c  = np & 15;
    int g  = (np >> 4) & 3;
    int jt = np >> 6;
    Ut2[i] = (__bf16)U[(size_t)k * (4 * HN) + g * HN + jt * 16 + c];
}

__global__ void k_zero_u32(unsigned* p) { *p = 0u; }

// ---------------------------------------------------------------------------
// Layer-0 input: [B,T,F] f32 -> time-major masked bf16 [t*64+b][F]
// ---------------------------------------------------------------------------
__global__ void k_premask(const float* __restrict__ x, const float* __restrict__ m,
                          __bf16* __restrict__ out) {
    constexpr int B_ = 64, T_ = 512, F_ = 128;
    int i = blockIdx.x * blockDim.x + threadIdx.x;
    if (i >= B_ * T_ * F_) return;
    int k = i % F_;
    int row = i / F_;              // t*64 + b
    int b = row & 63, t = row >> 6;
    out[i] = (__bf16)(x[((size_t)b * T_ + t) * F_ + k] * m[b * F_ + k]);
}

// ---------------------------------------------------------------------------
// Z_frag = Xbf[M,K] @ Wt^T + bias. One wave: 16-row tile x 4 col tiles
// (A-fragment reused across 4 WMMAs). Output in C-fragment order:
//   Z[(mt*NT + nt)*256 + r*32 + lane]
// ---------------------------------------------------------------------------
template <int K, int N>
__global__ void k_proj(const __bf16* __restrict__ Xbf,  // [M,K], row = t*64+b
                       const __bf16* __restrict__ Wt,   // [N,K]
                       const float* __restrict__ bias,  // [N]
                       float* __restrict__ Z, int M) {
    constexpr int NT = N / 16;
    constexpr int NG = N / 64;
    const int lane = threadIdx.x & 31;
    const int wave = threadIdx.x >> 5;
    const int job = blockIdx.x * (blockDim.x >> 5) + wave;
    if (job >= (M / 16) * NG) return;
    const int mt  = job / NG;
    const int nt0 = (job % NG) * 4;

    const int akb = (lane < 16) ? 0 : 8;     // A K-interleave base
    const int bkb = (lane < 16) ? 0 : 8;     // B K-interleave base
    const __bf16* arow = Xbf + (size_t)(mt * 16 + (lane & 15)) * K;

    v8f acc[4] = {};
    for (int kk = 0; kk < K; kk += 32) {
        Frag16 a;
        a.q[0] = *(const uint4*)(arow + kk + akb);
        a.q[1] = *(const uint4*)(arow + kk + akb + 16);
#pragma unroll
        for (int j = 0; j < 4; ++j) {
            const __bf16* brow =
                Wt + (size_t)((nt0 + j) * 16 + (lane & 15)) * K + kk + bkb;
            Frag16 b;
            b.q[0] = *(const uint4*)(brow);
            b.q[1] = *(const uint4*)(brow + 16);
            acc[j] = wmma_bf16(a.v, b.v, acc[j]);
        }
    }
#pragma unroll
    for (int j = 0; j < 4; ++j) {
        const int nt = nt0 + j;
        const float bv = bias[nt * 16 + (lane & 15)];
        float* zt = Z + ((size_t)mt * NT + nt) * 256 + lane;
#pragma unroll
        for (int r = 0; r < 8; ++r) zt[r * 32] = acc[j][r] + bv;
    }
}

// ---------------------------------------------------------------------------
// Persistent LSTM recurrence. Wave owns 16x16 (batch x hidden) block; cell
// state c in VGPRs for all T steps. Prologue: TDM-load this WG's 128-row
// slice of Ut2 into LDS (constant over time). Per step:
//   TDM: hbuf(global, double-buffered) -> LDS
//   z   = xz fragment tiles ; z += h @ U (WMMA, A and B from LDS)
//   gates -> c,h ; h -> global bf16 ; h*mask_next -> hs bf16
// ---------------------------------------------------------------------------
template <int HN>
__global__ void k_lstm(const float* __restrict__ xz,    // fragment tiles
                       const __bf16* __restrict__ Ut2,  // [4*HN, HN] reordered
                       const float* __restrict__ nmask, // [B,HN] next mask or null
                       __bf16* __restrict__ hs,         // [T*B, HN] or null
                       float* __restrict__ hfinal,      // [B,HN] or null
                       __bf16* __restrict__ hbuf,       // [2][B,HN]
                       unsigned* __restrict__ bar, int nwg) {
    constexpr int B_ = 64, T_ = 512;
    constexpr int NT = 4 * HN / 16;          // gate-col tiles per step
    constexpr int JT = HN / 16;
    __shared__ __bf16 hlds[B_ * HN];         // h(t-1), whole matrix
    __shared__ __bf16 utlds[128 * HN];       // this WG's U slice (2 jt x 4 g x 16)

    const int lane = threadIdx.x & 31;
    const int wave = threadIdx.x >> 5;
    const int wg   = blockIdx.x * (blockDim.x >> 5) + wave;
    const int rt   = wg & 3;                 // batch tile
    const int jt   = wg >> 2;                // hidden tile
    const int lj   = wave >> 2;              // jt within this WG (0/1)

    const int akb   = (lane < 16) ? 0 : 8;
    const int bkb   = (lane < 16) ? 0 : 8;
    const int arow  = rt * 16 + (lane & 15);         // batch row (A)
    const int ncol  = jt * 16 + (lane & 15);         // hidden col (B/C)
    const int crow0 = rt * 16 + ((lane < 16) ? 0 : 8);

    // --- TDM descriptors
    v4u g0h[2], g0u;
    v8i g1h, g1u;
#pragma unroll
    for (int p = 0; p < 2; ++p)
        tdm_desc(g0h[p], g1h, (unsigned)(uintptr_t)(&hlds[0]),
                 hbuf + (size_t)p * B_ * HN, HN, B_);
    tdm_desc(g0u, g1u, (unsigned)(uintptr_t)(&utlds[0]),
             Ut2 + (size_t)blockIdx.x * 128 * HN, HN, 128);

    // park this WG's recurrent-weight slice in LDS (once)
    if (wave == 0) tdm_load(g0u, g1u);

    // h(-1) = 0 in parity-0 buffer
#pragma unroll
    for (int r = 0; r < 8; ++r)
        hbuf[(size_t)(crow0 + r) * HN + ncol] = (__bf16)0.0f;

    // next-layer dropout mask for my block (constant over time)
    float nmv[8];
    if (hs) {
#pragma unroll
        for (int r = 0; r < 8; ++r) nmv[r] = nmask[(crow0 + r) * HN + ncol];
    }

    // B-fragment base in LDS: row (lj*64 + g*16 + col)
    const __bf16* ub0 = utlds + ((size_t)(lj * 64 + (lane & 15))) * HN + bkb;

    v8f c = {};
    gridbar(bar, (unsigned)nwg);

    for (int t = 0; t < T_; ++t) {
        // async-copy h(t-1) into LDS; one TDM op per workgroup
        if (wave == 0) {
            tdm_load(g0h[t & 1], g1h);
            __builtin_amdgcn_s_wait_tensorcnt(0);
        }
        __syncthreads();

        // z := xz fragment tiles for the 4 gates
        v8f z[4];
        const int mt = t * 4 + rt;
#pragma unroll
        for (int g = 0; g < 4; ++g) {
            const float* zt = xz + ((size_t)mt * NT + g * JT + jt) * 256 + lane;
#pragma unroll
            for (int r = 0; r < 8; ++r) z[g][r] = zt[r * 32];
        }

        // z += h @ U  (A and B from LDS)
        const __bf16* hp = hlds + (size_t)arow * HN;
#pragma unroll
        for (int kk = 0; kk < HN; kk += 32) {
            Frag16 a;
            a.q[0] = *(const uint4*)(hp + kk + akb);
            a.q[1] = *(const uint4*)(hp + kk + akb + 16);
#pragma unroll
            for (int g = 0; g < 4; ++g) {
                const __bf16* brow = ub0 + (size_t)g * 16 * HN + kk;
                Frag16 b;
                b.q[0] = *(const uint4*)(brow);
                b.q[1] = *(const uint4*)(brow + 16);
                z[g] = wmma_bf16(a.v, b.v, z[g]);
            }
        }

        // gates + state update
        v8f hnew;
#pragma unroll
        for (int r = 0; r < 8; ++r) {
            float iv = sigmoidf_(z[0][r]);
            float fv = sigmoidf_(z[1][r]);
            float gv = tanhf(z[2][r]);
            float ov = sigmoidf_(z[3][r]);
            float cv = fv * c[r] + iv * gv;
            c[r] = cv;
            hnew[r] = ov * tanhf(cv);
        }

        // publish h(t) into the other parity buffer; emit hs (masked bf16)
        __bf16* hb = hbuf + (size_t)((t + 1) & 1) * B_ * HN;
#pragma unroll
        for (int r = 0; r < 8; ++r) {
            const int brow = crow0 + r;
            hb[(size_t)brow * HN + ncol] = (__bf16)hnew[r];
            if (hs)
                hs[((size_t)t * B_ + brow) * HN + ncol] = (__bf16)(hnew[r] * nmv[r]);
        }
        if (hfinal && t == T_ - 1) {
#pragma unroll
            for (int r = 0; r < 8; ++r)
                hfinal[(size_t)(crow0 + r) * HN + ncol] = hnew[r];
        }
        gridbar(bar, (unsigned)((t + 2) * nwg));
    }
}

// ---------------------------------------------------------------------------
extern "C" void kernel_launch(void* const* d_in, const int* in_sizes, int n_in,
                              void* d_out, int out_size, void* d_ws, size_t ws_size,
                              hipStream_t stream) {
    (void)in_sizes; (void)n_in; (void)out_size; (void)ws_size;
    constexpr int B = 64, T = 512, F = 128, H = 256, L = 128;
    constexpr int M = B * T;

    const float* x  = (const float*)d_in[0];
    const float* W0 = (const float*)d_in[1];
    const float* U0 = (const float*)d_in[2];
    const float* b0 = (const float*)d_in[3];
    const float* W1 = (const float*)d_in[4];
    const float* U1 = (const float*)d_in[5];
    const float* b1 = (const float*)d_in[6];
    const float* W2 = (const float*)d_in[7];
    const float* U2 = (const float*)d_in[8];
    const float* b2 = (const float*)d_in[9];
    const float* m0 = (const float*)d_in[10];
    const float* m1 = (const float*)d_in[11];
    const float* m2 = (const float*)d_in[12];

    // workspace layout
    char* ws = (char*)d_ws;
    const size_t XZ_OFF  = 0;                       // 32768*1024 f32 = 128 MiB
    const size_t XBF_OFF = XZ_OFF  + (size_t)M * 1024 * 4;
    const size_t HSA_OFF = XBF_OFF + (size_t)M * F * 2;     // 8 MiB
    const size_t HSB_OFF = HSA_OFF + (size_t)M * H * 2;     // 16 MiB
    const size_t WT_OFF  = HSB_OFF + (size_t)M * H * 2;     // 16 MiB
    const size_t UT_OFF  = WT_OFF  + (1u << 20);
    const size_t HB_OFF  = UT_OFF  + (1u << 20);
    const size_t BAR_OFF = HB_OFF  + (1u << 17);

    float*    xzbuf = (float*)(ws + XZ_OFF);
    __bf16*   xbf   = (__bf16*)(ws + XBF_OFF);
    __bf16*   hs_a  = (__bf16*)(ws + HSA_OFF);
    __bf16*   hs_b  = (__bf16*)(ws + HSB_OFF);
    __bf16*   wt    = (__bf16*)(ws + WT_OFF);
    __bf16*   ut    = (__bf16*)(ws + UT_OFF);
    __bf16*   hbuf  = (__bf16*)(ws + HB_OFF);
    unsigned* bar   = (unsigned*)(ws + BAR_OFF);
    float*    out   = (float*)d_out;

    auto transp = [&](const float* w, __bf16* dst, int K, int N) {
        int n = K * N;
        k_transpose_bf16<<<(n + 255) / 256, 256, 0, stream>>>(w, dst, K, N);
    };
    auto transp_ut = [&](const float* u, __bf16* dst, int HN) {
        int n = HN * 4 * HN;
        k_transpose_ut<<<(n + 255) / 256, 256, 0, stream>>>(u, dst, HN);
    };

    // ---- layer 0: F=128 -> H=256 ----
    k_premask<<<(B * T * F + 255) / 256, 256, 0, stream>>>(x, m0, xbf);
    transp(W0, wt, F, 4 * H);
    transp_ut(U0, ut, H);
    k_proj<F, 4 * H><<<(M / 16) * (4 * H / 64) / 8, 256, 0, stream>>>(xbf, wt, b0, xzbuf, M);
    k_zero_u32<<<1, 1, 0, stream>>>(bar);
    k_lstm<H><<<8, 256, 0, stream>>>(xzbuf, ut, m1, hs_a, nullptr, hbuf, bar, 8);

    // ---- layer 1: H=256 -> H=256 ----
    transp(W1, wt, H, 4 * H);
    transp_ut(U1, ut, H);
    k_proj<H, 4 * H><<<(M / 16) * (4 * H / 64) / 8, 256, 0, stream>>>(hs_a, wt, b1, xzbuf, M);
    k_zero_u32<<<1, 1, 0, stream>>>(bar);
    k_lstm<H><<<8, 256, 0, stream>>>(xzbuf, ut, m2, hs_b, nullptr, hbuf, bar, 8);

    // ---- layer 2: H=256 -> L=128, final h only ----
    transp(W2, wt, H, 4 * L);
    transp_ut(U2, ut, L);
    k_proj<H, 4 * L><<<(M / 16) * (4 * L / 64) / 8, 256, 0, stream>>>(hs_b, wt, b2, xzbuf, M);
    k_zero_u32<<<1, 1, 0, stream>>>(bar);
    k_lstm<L><<<4, 256, 0, stream>>>(xzbuf, ut, nullptr, nullptr, out, hbuf, bar, 4);
}
